// TransformerBlock_15204184227910
// MI455X (gfx1250) — compile-verified, matrix operand order checked by reading, not run on hardware
//
#include <hip/hip_runtime.h>
#include <hip/hip_bf16.h>

typedef __attribute__((ext_vector_type(16))) _Float16 v16h;
typedef __attribute__((ext_vector_type(8)))  float    v8f;

#define EPS 1e-5f

// ---------------------------------------------------------------------------
// small utility kernels
// ---------------------------------------------------------------------------
__global__ void zero_kernel(float* p, int n) {
    int i = blockIdx.x * blockDim.x + threadIdx.x;
    if (i < n) p[i] = 0.f;
}

// Per-column sum / sumsq over a [rows x cols] row-major matrix (cols <= 64).
__global__ void colstats_kernel(const float* __restrict__ X, int rows, int cols,
                                float* __restrict__ sum, float* __restrict__ sumsq) {
    __shared__ float ls[256], lq[256];
    int t = threadIdx.x;
    int per  = 256 / cols;
    int used = per * cols;
    float s = 0.f, q = 0.f;
    if (t < used) {
        int col = t % cols;
        int r = blockIdx.x * per + t / cols;
        int stride = gridDim.x * per;
        for (; r < rows; r += stride) {
            float v = X[r * cols + col];
            s += v; q += v * v;
        }
    }
    ls[t] = s; lq[t] = q;
    __syncthreads();
    if (t < cols) {
        float S = 0.f, Q = 0.f;
        for (int i = t; i < used; i += cols) { S += ls[i]; Q += lq[i]; }
        atomicAdd(&sum[t], S);
        atomicAdd(&sumsq[t], Q);
    }
}

// scale = g * rsqrt(var + eps); shift = b - mean*scale   (biased var, matches jnp.var)
__global__ void finalize_bn_kernel(const float* __restrict__ sum, const float* __restrict__ sq,
                                   float inv_count,
                                   const float* __restrict__ g, const float* __restrict__ b,
                                   float* __restrict__ scale, float* __restrict__ shift, int cols) {
    int t = threadIdx.x;
    if (t < cols) {
        float m   = sum[t] * inv_count;
        float var = sq[t] * inv_count - m * m;
        float sc  = g[t] * rsqrtf(var + EPS);
        scale[t] = sc;
        shift[t] = b[t] - m * sc;
    }
}

// Y = [relu]( X*scale[col] + shift[col] (+ skip) )
__global__ void bn_apply_kernel(const float* __restrict__ X,
                                const float* __restrict__ scale, const float* __restrict__ shift,
                                const float* __restrict__ skip, float* __restrict__ Y,
                                int total, int cols, int do_relu) {
    int i = blockIdx.x * blockDim.x + threadIdx.x;
    if (i >= total) return;
    int c = i % cols;
    float v = X[i] * scale[c] + shift[c];
    if (skip) v += skip[i];
    if (do_relu) v = fmaxf(v, 0.f);
    Y[i] = v;
}

// ---------------------------------------------------------------------------
// WMMA f16 GEMM:  C[rows x 64] = A[rows x 64] @ W[64 x 64]
// W pre-packed into 8 fragments (2 k-steps x 4 n-tiles), each 32 lanes x 16 halves.
// Fragment layout (A, per ISA table; B mirrored with N in place of M):
//   lane L (L<16):  M/N = L, halves 0..7 = K k0+0..7,  halves 8..15 = K k0+16..23
//   lane L (L>=16): M/N = L-16, halves 0..7 = K k0+8..15, halves 8..15 = K k0+24..31
// ---------------------------------------------------------------------------
__global__ void pack_w64_kernel(const float* __restrict__ W, _Float16* __restrict__ out) {
    int t = threadIdx.x;            // 0..255 = 8 fragments x 32 lanes
    int frag = t >> 5;
    int lane = t & 31;
    int k0 = (frag >> 2) * 32;
    int j  = frag & 3;
    int n  = j * 16 + (lane & 15);
    int kb = k0 + ((lane >> 4) ? 8 : 0);
    _Float16* o = out + frag * 512 + lane * 16;
#pragma unroll
    for (int i = 0; i < 8; ++i) o[i]     = (_Float16)W[(kb + i) * 64 + n];
#pragma unroll
    for (int i = 0; i < 8; ++i) o[8 + i] = (_Float16)W[(kb + 16 + i) * 64 + n];
}

__device__ __forceinline__ void load_afrag(const float* __restrict__ ap, v16h& a) {
#pragma unroll
    for (int i = 0; i < 8; ++i) a[i]     = (_Float16)ap[i];
#pragma unroll
    for (int i = 0; i < 8; ++i) a[8 + i] = (_Float16)ap[16 + i];
}

// Each wave computes a 32x64 output block (two 16-row tiles sharing B fragments).
// row0 is forced into an SGPR via readfirstlane so every row-range branch lowers
// to a scalar branch: EXEC is all-ones around every WMMA (ISA requirement).
__global__ void gemm_n64_wmma_kernel(const float* __restrict__ A,
                                     const _Float16* __restrict__ packB,
                                     float* __restrict__ C, int rows) {
    int lane = threadIdx.x & 31;
    int wave = blockIdx.x * 8 + (threadIdx.x >> 5);
    int row0 = __builtin_amdgcn_readfirstlane(wave * 32);   // wave-uniform -> SGPR
    if (row0 >= rows) return;
    int m     = lane & 15;
    int koff  = (lane >> 4) * 8;
    int roff  = (lane >> 4) * 8;
    int cbase = lane & 15;

    if (row0 + 32 <= rows) {
        // -------- full path: straight line, no per-lane guards --------
        v8f acc0[4] = {};
        v8f acc1[4] = {};
        if (row0 + 32 < rows)
            __builtin_prefetch(A + (size_t)(row0 + 32) * 64, 0, 1);
#pragma unroll
        for (int ks = 0; ks < 2; ++ks) {
            int k0 = ks * 32;
            v16h a0, a1;
            load_afrag(A + (row0 + m) * 64 + k0 + koff, a0);
            load_afrag(A + (row0 + 16 + m) * 64 + k0 + koff, a1);
#pragma unroll
            for (int j = 0; j < 4; ++j) {
                const v16h b = *(const v16h*)(packB + (ks * 4 + j) * 512 + lane * 16);
                acc0[j] = __builtin_amdgcn_wmma_f32_16x16x32_f16(
                    false, a0, false, b, (short)0, acc0[j], false, false);
                acc1[j] = __builtin_amdgcn_wmma_f32_16x16x32_f16(
                    false, a1, false, b, (short)0, acc1[j], false, false);
            }
        }
        float* cp0 = C + (row0 + roff) * 64 + cbase;
        float* cp1 = C + (row0 + 16 + roff) * 64 + cbase;
#pragma unroll
        for (int j = 0; j < 4; ++j)
#pragma unroll
            for (int r = 0; r < 8; ++r) {
                cp0[r * 64 + j * 16] = acc0[j][r];
                cp1[r * 64 + j * 16] = acc1[j][r];
            }
    } else {
        // -------- tail path (rows not multiple of 32; never hit for N=40000) ----
        for (int tb = 0; tb < 2; ++tb) {
            int tbase = row0 + tb * 16;          // scalar
            if (tbase >= rows) break;            // scalar branch
            int rA = tbase + m;
            if (rA >= rows) rA = rows - 1;       // per-lane read clamp only
            v8f acc[4] = {};
#pragma unroll
            for (int ks = 0; ks < 2; ++ks) {
                v16h a;
                load_afrag(A + rA * 64 + ks * 32 + koff, a);
#pragma unroll
                for (int j = 0; j < 4; ++j) {
                    const v16h b = *(const v16h*)(packB + (ks * 4 + j) * 512 + lane * 16);
                    acc[j] = __builtin_amdgcn_wmma_f32_16x16x32_f16(
                        false, a, false, b, (short)0, acc[j], false, false);
                }
            }
#pragma unroll
            for (int j = 0; j < 4; ++j)
#pragma unroll
                for (int r = 0; r < 8; ++r) {
                    int rr = tbase + roff + r;
                    if (rr < rows) C[rr * 64 + j * 16 + cbase] = acc[j][r];
                }
        }
    }
}

// ---------------------------------------------------------------------------
// edge kernels
// ---------------------------------------------------------------------------
// t[e,:] = (pos[dst]-pos[src]) @ Wp1 + bp1        (E x 3)
__global__ void edge_pos_kernel(const float* __restrict__ pos,
                                const int* __restrict__ srcI, const int* __restrict__ dstI,
                                const float* __restrict__ Wp1, const float* __restrict__ bp1,
                                float* __restrict__ tOut, int Ee) {
    int e = blockIdx.x * blockDim.x + threadIdx.x;
    if (e >= Ee) return;
    int s = srcI[e], d = dstI[e];
    float rx = pos[d * 3 + 0] - pos[s * 3 + 0];
    float ry = pos[d * 3 + 1] - pos[s * 3 + 1];
    float rz = pos[d * 3 + 2] - pos[s * 3 + 2];
#pragma unroll
    for (int j = 0; j < 3; ++j)
        tOut[e * 3 + j] = rx * Wp1[0 * 3 + j] + ry * Wp1[1 * 3 + j] + rz * Wp1[2 * 3 + j] + bp1[j];
}

// column stats of a0 = a_dst[dst] - a_src[src] + delta  (recomputed, never stored)
__global__ void edge_a0_stats_kernel(const float* __restrict__ tIn,
                                     const float* __restrict__ scp, const float* __restrict__ shp,
                                     const float* __restrict__ adst, const float* __restrict__ asrc,
                                     const int* __restrict__ srcI, const int* __restrict__ dstI,
                                     const float* __restrict__ Wp2, const float* __restrict__ bp2,
                                     float* __restrict__ sum, float* __restrict__ sumsq, int Ee) {
    __shared__ float ls[256], lq[256];
    int t = threadIdx.x;
    int f = t & 63;
    int eg = t >> 6;                    // 0..3
    int base = blockIdx.x * 64;         // 64 edges per block
    float s = 0.f, q = 0.f;
    for (int i = 0; i < 16; ++i) {
        int e = base + eg + 4 * i;
        if (e < Ee) {
            float p0 = fmaxf(tIn[e * 3 + 0] * scp[0] + shp[0], 0.f);
            float p1 = fmaxf(tIn[e * 3 + 1] * scp[1] + shp[1], 0.f);
            float p2 = fmaxf(tIn[e * 3 + 2] * scp[2] + shp[2], 0.f);
            float delta = p0 * Wp2[f] + p1 * Wp2[64 + f] + p2 * Wp2[128 + f] + bp2[f];
            int sN = srcI[e], dN = dstI[e];
            float v = adst[dN * 64 + f] - asrc[sN * 64 + f] + delta;
            s += v; q += v * v;
        }
    }
    ls[t] = s; lq[t] = q;
    __syncthreads();
    if (t < 64) {
        atomicAdd(&sum[t],   ls[t] + ls[t + 64] + ls[t + 128] + ls[t + 192]);
        atomicAdd(&sumsq[t], lq[t] + lq[t + 64] + lq[t + 128] + lq[t + 192]);
    }
}

// z[e,:] = relu(bn(a0)) @ Wa1 + ba1     (E x 8), a0 recomputed on the fly
__global__ void edge_z_kernel(const float* __restrict__ tIn,
                              const float* __restrict__ scp, const float* __restrict__ shp,
                              const float* __restrict__ adst, const float* __restrict__ asrc,
                              const int* __restrict__ srcI, const int* __restrict__ dstI,
                              const float* __restrict__ Wp2, const float* __restrict__ bp2,
                              const float* __restrict__ sca, const float* __restrict__ sha,
                              const float* __restrict__ Wa1, const float* __restrict__ ba1,
                              float* __restrict__ z, int Ee) {
    __shared__ float w1[512], sS[64], hS[64], w2r[192], b2r[64], bS[8];
    int t = threadIdx.x;
    for (int i = t; i < 512; i += 256) w1[i] = Wa1[i];
    if (t < 64) { sS[t] = sca[t]; hS[t] = sha[t]; b2r[t] = bp2[t]; }
    if (t < 192) w2r[t] = Wp2[t];
    if (t < 8)  bS[t] = ba1[t];
    __syncthreads();
    int e = blockIdx.x * 32 + (t >> 3);
    if (e >= Ee) return;
    int j = t & 7;
    float p0 = fmaxf(tIn[e * 3 + 0] * scp[0] + shp[0], 0.f);
    float p1 = fmaxf(tIn[e * 3 + 1] * scp[1] + shp[1], 0.f);
    float p2 = fmaxf(tIn[e * 3 + 2] * scp[2] + shp[2], 0.f);
    int sN = srcI[e], dN = dstI[e];
    const float* rd = adst + dN * 64;
    const float* rs = asrc + sN * 64;
    float acc = bS[j];
    for (int k = 0; k < 64; ++k) {
        float delta = p0 * w2r[k] + p1 * w2r[64 + k] + p2 * w2r[128 + k] + b2r[k];
        float a0v = rd[k] - rs[k] + delta;
        float v = fmaxf(a0v * sS[k] + hS[k], 0.f);
        acc += v * w1[k * 8 + j];
    }
    z[e * 8 + j] = acc;
}

// Per-node: a = relu(bn(z)) @ Wa2 + ba2; per-feature softmax over the node's
// 16 contiguous edges; agg = sum attn * (v[src] + delta).
__global__ void edge_softmax_agg_kernel(const float* __restrict__ z,
                                        const float* __restrict__ scz, const float* __restrict__ shz,
                                        const float* __restrict__ Wa2, const float* __restrict__ ba2,
                                        const float* __restrict__ tIn,
                                        const float* __restrict__ scp, const float* __restrict__ shp,
                                        const float* __restrict__ Wp2, const float* __restrict__ bp2,
                                        const float* __restrict__ vM, const int* __restrict__ srcI,
                                        float* __restrict__ agg, int Nn) {
    __shared__ float w2[512];     // Wa2 [8][64]
    __shared__ float b2s[64], wp2s[192], bp2s[64];
    __shared__ float zl[512];     // relu(bn(z)): 4 nodes * 16 edges * 8
    __shared__ float phl[192];    // ph: 64 edges * 3
    __shared__ int   srcl[64];
    int t = threadIdx.x;
    int node0 = blockIdx.x * 4;
    int ebase = node0 * 16;
    for (int i = t; i < 512; i += 256) w2[i] = Wa2[i];
    if (t < 64) { b2s[t] = ba2[t]; bp2s[t] = bp2[t]; }
    if (t < 192) wp2s[t] = Wp2[t];
    for (int i = t; i < 512; i += 256) {
        int ee = i >> 3, j = i & 7;
        float v = (node0 + (ee >> 4) < Nn) ? z[(ebase + ee) * 8 + j] : 0.f;
        zl[i] = fmaxf(v * scz[j] + shz[j], 0.f);
    }
    for (int i = t; i < 192; i += 256) {
        int ee = i / 3, j = i % 3;
        float v = (node0 + (ee >> 4) < Nn) ? tIn[(ebase + ee) * 3 + j] : 0.f;
        phl[i] = fmaxf(v * scp[j] + shp[j], 0.f);
    }
    if (t < 64) srcl[t] = (node0 + (t >> 4) < Nn) ? srcI[ebase + t] : 0;
    __syncthreads();

    int ln = t >> 6;
    int node = node0 + ln;
    if (node >= Nn) return;
    int f = t & 63;

    float aval[16];
    float amax = -3.4e38f;
#pragma unroll
    for (int e = 0; e < 16; ++e) {
        const float* zr = &zl[(ln * 16 + e) * 8];
        float acc = b2s[f];
#pragma unroll
        for (int j = 0; j < 8; ++j) acc += zr[j] * w2[j * 64 + f];
        aval[e] = acc;
        amax = fmaxf(amax, acc);
    }
    float s = 0.f;
#pragma unroll
    for (int e = 0; e < 16; ++e) { aval[e] = expf(aval[e] - amax); s += aval[e]; }
    float inv = 1.f / (s + 1e-16f);
    float out = 0.f;
#pragma unroll
    for (int e = 0; e < 16; ++e) {
        const float* ph = &phl[(ln * 16 + e) * 3];
        float delta = ph[0] * wp2s[f] + ph[1] * wp2s[64 + f] + ph[2] * wp2s[128 + f] + bp2s[f];
        float vv = vM[srcl[ln * 16 + e] * 64 + f] + delta;
        out += aval[e] * inv * vv;
    }
    agg[node * 64 + f] = out;
}

// ---------------------------------------------------------------------------
extern "C" void kernel_launch(void* const* d_in, const int* in_sizes, int n_in,
                              void* d_out, int out_size, void* d_ws, size_t ws_size,
                              hipStream_t stream) {
    const float* x     = (const float*)d_in[0];
    const float* pos   = (const float*)d_in[1];
    const int*   eidx  = (const int*)  d_in[2];
    const float* W_in  = (const float*)d_in[3];
    const float* g1    = (const float*)d_in[4];
    const float* b1    = (const float*)d_in[5];
    const float* W_v   = (const float*)d_in[6];
    const float* W_src = (const float*)d_in[7];
    const float* W_dst = (const float*)d_in[8];
    const float* Wp1   = (const float*)d_in[9];
    const float* bp1   = (const float*)d_in[10];
    const float* gp    = (const float*)d_in[11];
    const float* bp    = (const float*)d_in[12];
    const float* Wp2   = (const float*)d_in[13];
    const float* bp2   = (const float*)d_in[14];
    const float* ga0   = (const float*)d_in[15];
    const float* ba0   = (const float*)d_in[16];
    const float* Wa1   = (const float*)d_in[17];
    const float* ba1   = (const float*)d_in[18];
    const float* ga1   = (const float*)d_in[19];
    const float* bb1   = (const float*)d_in[20];
    const float* Wa2   = (const float*)d_in[21];
    const float* ba2   = (const float*)d_in[22];
    const float* g2    = (const float*)d_in[23];
    const float* b2    = (const float*)d_in[24];
    const float* W_out = (const float*)d_in[25];
    const float* g3    = (const float*)d_in[26];
    const float* b3    = (const float*)d_in[27];

    const int Nn = in_sizes[0] / 64;
    const int Ee = in_sizes[2] / 2;
    const int* srcI = eidx;
    const int* dstI = eidx + Ee;
    float* out = (float*)d_out;

    // workspace carve-up (floats)
    float* ws = (float*)d_ws;
    // stats: 6 stages * (sum64 | sq64 | scale64 | shift64)
    float* st[6][4];
    for (int sIdx = 0; sIdx < 6; ++sIdx)
        for (int k = 0; k < 4; ++k) st[sIdx][k] = ws + sIdx * 256 + k * 64;
    _Float16* pw = (_Float16*)(ws + 2048);           // 5 * 4096 halves
    const int Nf = Nn * 64;
    float* hBuf = ws + 16384;
    float* asrc = hBuf + Nf;
    float* adst = asrc + Nf;
    float* vM   = adst + Nf;
    float* tBuf = vM + Nf;          // E*3
    float* zBuf = tBuf + 3 * Ee;    // E*8
    float* agg  = zBuf + 8 * (size_t)Ee;  // N*64
    float* y3   = agg + Nf;               // N*64

    const int gemmGrid = ((Nn + 31) / 32 + 7) / 8;   // 32 rows per wave, 8 waves per block
    const float invN = 1.0f / (float)Nn;
    const float invE = 1.0f / (float)Ee;

    // 0) zero stat accumulators
    zero_kernel<<<6, 256, 0, stream>>>(ws, 1536);

    // 1) pack weights into WMMA fragments
    pack_w64_kernel<<<1, 256, 0, stream>>>(W_in,  pw);
    pack_w64_kernel<<<1, 256, 0, stream>>>(W_src, pw + 4096);
    pack_w64_kernel<<<1, 256, 0, stream>>>(W_dst, pw + 2 * 4096);
    pack_w64_kernel<<<1, 256, 0, stream>>>(W_v,   pw + 3 * 4096);
    pack_w64_kernel<<<1, 256, 0, stream>>>(W_out, pw + 4 * 4096);

    // 2) h = relu(bn(x @ W_in))
    gemm_n64_wmma_kernel<<<gemmGrid, 256, 0, stream>>>(x, pw, hBuf, Nn);
    colstats_kernel<<<512, 256, 0, stream>>>(hBuf, Nn, 64, st[0][0], st[0][1]);
    finalize_bn_kernel<<<1, 64, 0, stream>>>(st[0][0], st[0][1], invN, g1, b1, st[0][2], st[0][3], 64);
    bn_apply_kernel<<<(Nf + 255) / 256, 256, 0, stream>>>(hBuf, st[0][2], st[0][3], nullptr, hBuf, Nf, 64, 1);

    // 3) a_src, a_dst, v projections
    gemm_n64_wmma_kernel<<<gemmGrid, 256, 0, stream>>>(hBuf, pw + 4096,     asrc, Nn);
    gemm_n64_wmma_kernel<<<gemmGrid, 256, 0, stream>>>(hBuf, pw + 2 * 4096, adst, Nn);
    gemm_n64_wmma_kernel<<<gemmGrid, 256, 0, stream>>>(hBuf, pw + 3 * 4096, vM,   Nn);

    // 4) edge geometry: t = rel @ Wp1 + bp1, then BN stats (cols=3)
    edge_pos_kernel<<<(Ee + 255) / 256, 256, 0, stream>>>(pos, srcI, dstI, Wp1, bp1, tBuf, Ee);
    colstats_kernel<<<1024, 256, 0, stream>>>(tBuf, Ee, 3, st[1][0], st[1][1]);
    finalize_bn_kernel<<<1, 64, 0, stream>>>(st[1][0], st[1][1], invE, gp, bp, st[1][2], st[1][3], 3);

    // 5) stats of a0 (recomputed, not stored)
    edge_a0_stats_kernel<<<(Ee + 63) / 64, 256, 0, stream>>>(
        tBuf, st[1][2], st[1][3], adst, asrc, srcI, dstI, Wp2, bp2, st[2][0], st[2][1], Ee);
    finalize_bn_kernel<<<1, 64, 0, stream>>>(st[2][0], st[2][1], invE, ga0, ba0, st[2][2], st[2][3], 64);

    // 6) z = relu(bn(a0)) @ Wa1 + ba1, then BN stats (cols=8)
    edge_z_kernel<<<(Ee + 31) / 32, 256, 0, stream>>>(
        tBuf, st[1][2], st[1][3], adst, asrc, srcI, dstI, Wp2, bp2,
        st[2][2], st[2][3], Wa1, ba1, zBuf, Ee);
    colstats_kernel<<<1024, 256, 0, stream>>>(zBuf, Ee, 8, st[3][0], st[3][1]);
    finalize_bn_kernel<<<1, 64, 0, stream>>>(st[3][0], st[3][1], invE, ga1, bb1, st[3][2], st[3][3], 8);

    // 7) per-node softmax + aggregation
    edge_softmax_agg_kernel<<<(Nn + 3) / 4, 256, 0, stream>>>(
        zBuf, st[3][2], st[3][3], Wa2, ba2, tBuf, st[1][2], st[1][3],
        Wp2, bp2, vM, srcI, agg, Nn);

    // 8) h2 = relu(bn(agg)); y3 = h2 @ W_out
    colstats_kernel<<<512, 256, 0, stream>>>(agg, Nn, 64, st[4][0], st[4][1]);
    finalize_bn_kernel<<<1, 64, 0, stream>>>(st[4][0], st[4][1], invN, g2, b2, st[4][2], st[4][3], 64);
    bn_apply_kernel<<<(Nf + 255) / 256, 256, 0, stream>>>(agg, st[4][2], st[4][3], nullptr, agg, Nf, 64, 1);
    gemm_n64_wmma_kernel<<<gemmGrid, 256, 0, stream>>>(agg, pw + 4 * 4096, y3, Nn);

    // 9) out = relu(bn(y3) + x)
    colstats_kernel<<<512, 256, 0, stream>>>(y3, Nn, 64, st[5][0], st[5][1]);
    finalize_bn_kernel<<<1, 64, 0, stream>>>(st[5][0], st[5][1], invN, g3, b3, st[5][2], st[5][3], 64);
    bn_apply_kernel<<<(Nf + 255) / 256, 256, 0, stream>>>(y3, st[5][2], st[5][3], x, out, Nf, 64, 1);
}
